// TransformerDecoderLayer_14877766713599
// MI455X (gfx1250) — matrix operand order repaired and checked by reading, not compiled, not run on hardware
//
#include <hip/hip_runtime.h>

// ---------------------------------------------------------------------------
// Transformer decoder layer for MI455X (gfx1250).
// bf16 WMMA GEMMs (f32 accumulate), TDM bf16 tile staging, one-time f32->bf16
// conversion so the GEMM inner loops are pure ds_load_b128 -> v_wmma.
// E=1024, H=16, HD=64, FF=4096, T=S=1024, B=4
// ---------------------------------------------------------------------------

typedef __attribute__((ext_vector_type(16))) __bf16 v16bf;
typedef __attribute__((ext_vector_type(4)))  __bf16 v4bf;
typedef __attribute__((ext_vector_type(8)))  float  v8f;
typedef __attribute__((ext_vector_type(4)))  unsigned int v4u;
typedef __attribute__((ext_vector_type(8)))  int v8i;
typedef __attribute__((ext_vector_type(4)))  int v4i;

#define E_DIM  1024
#define H_NUM  16
#define HD_DIM 64
#define FF_DIM 4096
#define T_DIM  1024
#define S_DIM  1024
#define B_DIM  4

__device__ __forceinline__ __bf16 f2bf(float x) { return (__bf16)x; }

__device__ __forceinline__ v8f wmma_bf16(v16bf a, v16bf b, v8f c) {
  // D = A(16x32) * B(32x16) + C, f32 accumulate
  return __builtin_amdgcn_wmma_f32_16x16x32_bf16(false, a, false, b,
                                                 (short)0, c, false, false);
}

// ---------------------------------------------------------------------------
// TDM: DMA a 2D bf16 tile [rows x cols] from global (row stride = strideElems
// bf16 elements) into LDS at byte address lds_addr (contiguous rows*cols*2 B).
// D# per CDNA5 ISA; groups 2/3 zeroed (2D tile).
// ---------------------------------------------------------------------------
__device__ __forceinline__ void tdm_load_2d_bf16(unsigned lds_addr, const __bf16* gptr,
                                                 int rows, int cols, int strideElems,
                                                 int tensorRows) {
  unsigned long long ga = (unsigned long long)gptr;
  v4u g0;
  g0[0] = 1u;                                          // count=1, user D#
  g0[1] = lds_addr;                                    // LDS byte address
  g0[2] = (unsigned)(ga & 0xFFFFFFFFull);              // global_addr[31:0]
  g0[3] = (unsigned)((ga >> 32) & 0x01FFFFFFull)       // global_addr[56:32]
          | (2u << 30);                                // type = 2 ("image")
  v8i g1;
  g1[0] = (int)(1u << 16);                             // data_size = 2 bytes
  g1[1] = (int)(((unsigned)strideElems & 0xFFFFu) << 16);        // tensor_dim0 lo16
  g1[2] = (int)((((unsigned)strideElems >> 16) & 0xFFFFu)        // tensor_dim0 hi16
          | (((unsigned)tensorRows & 0xFFFFu) << 16));           // tensor_dim1 lo16
  g1[3] = (int)((((unsigned)tensorRows >> 16) & 0xFFFFu)         // tensor_dim1 hi16
          | (((unsigned)cols & 0xFFFFu) << 16));                 // tile_dim0
  g1[4] = (int)((unsigned)rows & 0xFFFFu);                       // tile_dim1 (tile_dim2=0)
  g1[5] = (int)(unsigned)strideElems;                            // tensor_dim0_stride[31:0]
  g1[6] = 0;
  g1[7] = 0;
  v4i z4 = {};
#if defined(__clang_major__) && (__clang_major__ >= 23)
  v8i z8 = {};
  __builtin_amdgcn_tensor_load_to_lds(g0, g1, z4, z4, z8, 0);
#else
  __builtin_amdgcn_tensor_load_to_lds(g0, g1, z4, z4, 0);
#endif
}

// ---------------------------------------------------------------------------
// One-time f32 -> bf16 conversion (vectorized, grid-stride). n % 4 == 0.
// ---------------------------------------------------------------------------
__global__ __launch_bounds__(256)
void cvt_bf16_kernel(const float* __restrict__ src, __bf16* __restrict__ dst, int n) {
  int n4 = n >> 2;
  for (int i = blockIdx.x * 256 + threadIdx.x; i < n4; i += gridDim.x * 256) {
    float4 f = ((const float4*)src)[i];
    v4bf o;
    o[0] = f2bf(f.x); o[1] = f2bf(f.y); o[2] = f2bf(f.z); o[3] = f2bf(f.w);
    ((v4bf*)dst)[i] = o;
  }
}

// ---------------------------------------------------------------------------
// GEMM: C[M,N] = act(A[M,K] @ W[N,K]^T + bias[N]);  M,N % 64 == 0, K % 32 == 0
// A, W are bf16. 128 threads = 4 waves; 64x64 workgroup tile; TDM-staged bf16
// tiles in LDS; inner loop is ds_load_b128 -> v_wmma with no conversions.
// Output: f32 to Cf (if non-null) and/or bf16 to Cb (if non-null).
// ---------------------------------------------------------------------------
#define GT_K 32

__global__ __launch_bounds__(128)
void gemm_bias_kernel(const __bf16* __restrict__ A, const __bf16* __restrict__ W,
                      const float* __restrict__ bias,
                      float* __restrict__ Cf, __bf16* __restrict__ Cb,
                      int M, int N, int K, int relu) {
  __shared__ __bf16 Af[64 * GT_K];   // 4 KB bf16 A tile
  __shared__ __bf16 Wf[64 * GT_K];   // 4 KB bf16 W tile
  const int tid  = threadIdx.x;
  const int lane = tid & 31, wave = tid >> 5;
  const int mlane = lane & 15, hf = lane >> 4;
  const int m0 = blockIdx.x * 64;
  const int n0 = blockIdx.y * 64;

  const unsigned ldsA = (unsigned)(unsigned long long)(void*)Af;
  const unsigned ldsW = (unsigned)(unsigned long long)(void*)Wf;

  v8f acc[4] = {};

  for (int k0 = 0; k0 < K; k0 += GT_K) {
    // TDM stage both tiles (one issue per workgroup, from wave 0)
    if (wave == 0) {
      tdm_load_2d_bf16(ldsA, A + (size_t)m0 * K + k0, 64, GT_K, K, M);
      tdm_load_2d_bf16(ldsW, W + (size_t)n0 * K + k0, 64, GT_K, K, N);
      __builtin_amdgcn_s_wait_tensorcnt(0);
    }
    __syncthreads();

    // A fragment: lane holds row (wave*16 + mlane); K = hf*16 + e (contiguous)
    v16bf a;
    const __bf16* arow = &Af[(wave * 16 + mlane) * GT_K];
#pragma unroll
    for (int e = 0; e < 8; ++e) {
      a[e]     = arow[hf * 16 + e];
      a[e + 8] = arow[hf * 16 + 8 + e];
    }
#pragma unroll
    for (int j = 0; j < 4; ++j) {
      // B fragment: lane holds column (j*16 + mlane); K = hf*16 + e
      v16bf bfr;
      const __bf16* wrow = &Wf[(j * 16 + mlane) * GT_K + hf * 16];
#pragma unroll
      for (int e = 0; e < 16; ++e) bfr[e] = wrow[e];
      acc[j] = wmma_bf16(a, bfr, acc[j]);
    }
    __syncthreads();   // before next TDM overwrites the tiles
  }

  // epilogue: D fragment layout (VGPR r -> row hf*8+r, lane&15 -> col)
#pragma unroll
  for (int j = 0; j < 4; ++j) {
    int n = n0 + j * 16 + mlane;
    float bv = bias[n];
#pragma unroll
    for (int r = 0; r < 8; ++r) {
      int m = m0 + wave * 16 + hf * 8 + r;
      float val = acc[j][r] + bv;
      if (relu) val = fmaxf(val, 0.0f);
      if (Cf) Cf[(size_t)m * N + n] = val;
      if (Cb) Cb[(size_t)m * N + n] = f2bf(val);
    }
  }
}

// ---------------------------------------------------------------------------
// Attention: one workgroup per (b, h, 16 rows of T). q/k/v are bf16 [tok, E]
// with token = t*B + b. Full S=1024 score row in 64KB dynamic LDS (f32),
// exact softmax, P@V with bf16 WMMA. 1/sqrt(HD) applied on f32 accumulator.
// ctx written bf16 [T,B,E]; probs (optional) f32 [H,B,T,S].
// ---------------------------------------------------------------------------
__global__ __launch_bounds__(128)
void attn_kernel(const __bf16* __restrict__ q, const __bf16* __restrict__ k,
                 const __bf16* __restrict__ v, const float* __restrict__ mask,
                 __bf16* __restrict__ ctx, float* __restrict__ probs) {
  extern __shared__ float sc[];   // [16][S_DIM] f32 scores / probabilities
  const int tid  = threadIdx.x;
  const int lane = tid & 31, wave = tid >> 5;
  const int mlane = lane & 15, hf = lane >> 4;
  const int t0 = blockIdx.x * 16;
  const int bh = blockIdx.y;
  const int b  = bh / H_NUM, h = bh % H_NUM;

  // --- Q fragments (A-matrix, 16 x 64 over d), straight bf16 loads ---
  const __bf16* qrow = q + (size_t)((t0 + mlane) * B_DIM + b) * E_DIM + h * HD_DIM;
  v16bf qa0, qa1;
#pragma unroll
  for (int e = 0; e < 8; ++e) {
    qa0[e]     = qrow[hf * 16 + e];
    qa0[e + 8] = qrow[hf * 16 + 8 + e];
    qa1[e]     = qrow[32 + hf * 16 + e];
    qa1[e + 8] = qrow[32 + hf * 16 + 8 + e];
  }

  // --- phase 1: scores = (Q @ K^T) * 0.125 (+ mask) ---
  for (int scnk = wave; scnk < S_DIM / 16; scnk += 4) {
    const int s0 = scnk * 16;
    const __bf16* krow = k + (size_t)((s0 + mlane) * B_DIM + b) * E_DIM + h * HD_DIM;
    v16bf kb0, kb1;   // B-matrix: lane = column s, K = d contiguous
#pragma unroll
    for (int e = 0; e < 16; ++e) {
      kb0[e] = krow[hf * 16 + e];
      kb1[e] = krow[32 + hf * 16 + e];
    }
    v8f acc = {};
    acc = wmma_bf16(qa0, kb0, acc);
    acc = wmma_bf16(qa1, kb1, acc);
#pragma unroll
    for (int r = 0; r < 8; ++r) {
      int ml  = hf * 8 + r;
      int col = s0 + mlane;
      float val = acc[r] * 0.125f;
      if (mask) val += mask[(size_t)(t0 + ml) * S_DIM + col];
      sc[ml * S_DIM + col] = val;
    }
  }
  __syncthreads();

  // --- phase 2: row softmax (8 contiguous lanes per row, shfl reductions) ---
  {
    const int row = tid >> 3;   // 0..15
    const int sub = tid & 7;    // 0..7
    float mx = -3.0e38f;
    for (int c = sub; c < S_DIM; c += 8) mx = fmaxf(mx, sc[row * S_DIM + c]);
#pragma unroll
    for (int o = 4; o; o >>= 1) mx = fmaxf(mx, __shfl_xor(mx, o, 32));
    float sum = 0.0f;
    for (int c = sub; c < S_DIM; c += 8) {
      float e = __expf(sc[row * S_DIM + c] - mx);
      sc[row * S_DIM + c] = e;
      sum += e;
    }
#pragma unroll
    for (int o = 4; o; o >>= 1) sum += __shfl_xor(sum, o, 32);
    const float inv = 1.0f / sum;
    for (int c = sub; c < S_DIM; c += 8) {
      float p = sc[row * S_DIM + c] * inv;
      sc[row * S_DIM + c] = p;
      if (probs)
        probs[((size_t)(h * B_DIM + b) * T_DIM + (t0 + row)) * S_DIM + c] = p;
    }
  }
  __syncthreads();

  // --- phase 3: ctx(16x64) = P(16x1024) @ V(1024x64); wave owns 16 d-cols ---
  v8f oacc = {};
  for (int s0 = 0; s0 < S_DIM; s0 += 32) {
    v16bf pa;   // A-matrix from LDS scores (cvt needed: probs live as f32)
#pragma unroll
    for (int e = 0; e < 8; ++e) {
      pa[e]     = f2bf(sc[mlane * S_DIM + s0 + hf * 16 + e]);
      pa[e + 8] = f2bf(sc[mlane * S_DIM + s0 + hf * 16 + 8 + e]);
    }
    v16bf vb;   // B-matrix: lane = column d, K = s (bf16 direct)
    const int n = wave * 16 + mlane;
#pragma unroll
    for (int e = 0; e < 16; ++e) {
      int s = s0 + hf * 16 + e;
      vb[e] = v[(size_t)(s * B_DIM + b) * E_DIM + h * HD_DIM + n];
    }
    oacc = wmma_bf16(pa, vb, oacc);
  }
#pragma unroll
  for (int r = 0; r < 8; ++r) {
    int t = t0 + hf * 8 + r;
    ctx[(size_t)(t * B_DIM + b) * E_DIM + h * HD_DIM + wave * 16 + mlane] = f2bf(oacc[r]);
  }
}

// ---------------------------------------------------------------------------
// Fused residual add + LayerNorm; f32 output + optional bf16 shadow output.
// ---------------------------------------------------------------------------
__global__ __launch_bounds__(256)
void add_ln_kernel(const float* __restrict__ x, const float* __restrict__ res,
                   const float* __restrict__ g, const float* __restrict__ bta,
                   float* __restrict__ out, __bf16* __restrict__ out_bf) {
  const int rowi = blockIdx.x;
  const float* xr = x   + (size_t)rowi * E_DIM;
  const float* rr = res + (size_t)rowi * E_DIM;
  float sum = 0.0f, sq = 0.0f;
  for (int c = threadIdx.x; c < E_DIM; c += 256) {
    float vv = xr[c] + rr[c];
    sum += vv; sq += vv * vv;
  }
#pragma unroll
  for (int o = 16; o; o >>= 1) {
    sum += __shfl_xor(sum, o, 32);
    sq  += __shfl_xor(sq,  o, 32);
  }
  __shared__ float s1[8], s2[8];
  if ((threadIdx.x & 31) == 0) { s1[threadIdx.x >> 5] = sum; s2[threadIdx.x >> 5] = sq; }
  __syncthreads();
  sum = 0.0f; sq = 0.0f;
#pragma unroll
  for (int i = 0; i < 8; ++i) { sum += s1[i]; sq += s2[i]; }
  const float mu  = sum * (1.0f / E_DIM);
  const float var = sq * (1.0f / E_DIM) - mu * mu;
  const float inv = rsqrtf(var + 1e-5f);
  for (int c = threadIdx.x; c < E_DIM; c += 256) {
    float vv = xr[c] + rr[c];
    float o = (vv - mu) * inv * g[c] + bta[c];
    out[(size_t)rowi * E_DIM + c] = o;
    if (out_bf) out_bf[(size_t)rowi * E_DIM + c] = f2bf(o);
  }
}

// ---------------------------------------------------------------------------
// Host orchestration
// ---------------------------------------------------------------------------
extern "C" void kernel_launch(void* const* d_in, const int* in_sizes, int n_in,
                              void* d_out, int out_size, void* d_ws, size_t ws_size,
                              hipStream_t stream) {
  (void)in_sizes; (void)n_in; (void)out_size; (void)ws_size;

  const float* state = (const float*)d_in[0];   // [T,B,E]
  const float* enc   = (const float*)d_in[1];   // [S,B,E]
  const float* mask  = (const float*)d_in[2];   // [T,T]
  const float* sa_q_w = (const float*)d_in[3];  const float* sa_q_b = (const float*)d_in[4];
  const float* sa_k_w = (const float*)d_in[5];  const float* sa_k_b = (const float*)d_in[6];
  const float* sa_v_w = (const float*)d_in[7];  const float* sa_v_b = (const float*)d_in[8];
  const float* sa_o_w = (const float*)d_in[9];  const float* sa_o_b = (const float*)d_in[10];
  const float* ca_q_w = (const float*)d_in[11]; const float* ca_q_b = (const float*)d_in[12];
  const float* ca_k_w = (const float*)d_in[13]; const float* ca_k_b = (const float*)d_in[14];
  const float* ca_v_w = (const float*)d_in[15]; const float* ca_v_b = (const float*)d_in[16];
  const float* ca_o_w = (const float*)d_in[17]; const float* ca_o_b = (const float*)d_in[18];
  const float* ln1_g = (const float*)d_in[19];  const float* ln1_b = (const float*)d_in[20];
  const float* ln2_g = (const float*)d_in[21];  const float* ln2_b = (const float*)d_in[22];
  const float* ln3_g = (const float*)d_in[23];  const float* ln3_b = (const float*)d_in[24];
  const float* fc1_w = (const float*)d_in[25];  const float* fc1_b = (const float*)d_in[26];
  const float* fc2_w = (const float*)d_in[27];  const float* fc2_b = (const float*)d_in[28];

  float* out_z    = (float*)d_out;                                   // [T,B,E]
  float* out_attn = (float*)d_out + (size_t)T_DIM * B_DIM * E_DIM;   // [H,B,T,S]

  const int MTOK = T_DIM * B_DIM;              // 4096 token rows
  const size_t NTOKE = (size_t)MTOK * E_DIM;   // 4M elements
  const size_t NWEE  = (size_t)E_DIM * E_DIM;  // 1M elements
  const size_t NWFE  = (size_t)FF_DIM * E_DIM; // 4M elements

  // ---- workspace carve (byte offsets; every size is MB-aligned) ----
  char* wp = (char*)d_ws;
  auto carve = [&](size_t bytes) { char* p = wp; wp += bytes; return p; };
  __bf16* state_bf = (__bf16*)carve(NTOKE * 2);
  __bf16* enc_bf   = (__bf16*)carve(NTOKE * 2);
  __bf16* w_saq = (__bf16*)carve(NWEE * 2);
  __bf16* w_sak = (__bf16*)carve(NWEE * 2);
  __bf16* w_sav = (__bf16*)carve(NWEE * 2);
  __bf16* w_sao = (__bf16*)carve(NWEE * 2);
  __bf16* w_caq = (__bf16*)carve(NWEE * 2);
  __bf16* w_cak = (__bf16*)carve(NWEE * 2);
  __bf16* w_cav = (__bf16*)carve(NWEE * 2);
  __bf16* w_cao = (__bf16*)carve(NWEE * 2);
  __bf16* w_f1  = (__bf16*)carve(NWFE * 2);
  __bf16* w_f2  = (__bf16*)carve(NWFE * 2);
  __bf16* qb    = (__bf16*)carve(NTOKE * 2);
  __bf16* kb    = (__bf16*)carve(NTOKE * 2);
  __bf16* vb    = (__bf16*)carve(NTOKE * 2);
  __bf16* ctxb  = (__bf16*)carve(NTOKE * 2);
  __bf16* xb    = (__bf16*)carve(NTOKE * 2);
  __bf16* yb    = (__bf16*)carve(NTOKE * 2);
  __bf16* f1b   = (__bf16*)carve((size_t)MTOK * FF_DIM * 2);
  float*  tmp0  = (float*)carve(NTOKE * 4);
  float*  x32   = (float*)carve(NTOKE * 4);
  float*  y32   = (float*)carve(NTOKE * 4);

  const dim3 gblk(128);
  const dim3 agrid(T_DIM / 16, B_DIM * H_NUM);
  const size_t ashm = (size_t)16 * S_DIM * sizeof(float);  // 64 KB

  auto cvt = [&](const float* s, __bf16* d, size_t n) {
    int blocks = (int)((n / 4 + 255) / 256);
    if (blocks > 1024) blocks = 1024;
    cvt_bf16_kernel<<<blocks, 256, 0, stream>>>(s, d, (int)n);
  };
  auto gemm = [&](const __bf16* A, const __bf16* W, const float* bias,
                  float* Cf, __bf16* Cb, int M, int N, int K, int relu) {
    dim3 grid(M / 64, N / 64);
    gemm_bias_kernel<<<grid, gblk, 0, stream>>>(A, W, bias, Cf, Cb, M, N, K, relu);
  };

  // ---- one-time bf16 conversion of inputs and weights ----
  cvt(state, state_bf, NTOKE);
  cvt(enc,   enc_bf,   NTOKE);
  cvt(sa_q_w, w_saq, NWEE);  cvt(sa_k_w, w_sak, NWEE);
  cvt(sa_v_w, w_sav, NWEE);  cvt(sa_o_w, w_sao, NWEE);
  cvt(ca_q_w, w_caq, NWEE);  cvt(ca_k_w, w_cak, NWEE);
  cvt(ca_v_w, w_cav, NWEE);  cvt(ca_o_w, w_cao, NWEE);
  cvt(fc1_w, w_f1, NWFE);    cvt(fc2_w, w_f2, NWFE);

  // ---------------- self-attention block ----------------
  gemm(state_bf, w_saq, sa_q_b, nullptr, qb, MTOK, E_DIM, E_DIM, 0);   // Q (bf16)
  gemm(state_bf, w_sak, sa_k_b, nullptr, kb, MTOK, E_DIM, E_DIM, 0);   // K (bf16)
  gemm(state_bf, w_sav, sa_v_b, nullptr, vb, MTOK, E_DIM, E_DIM, 0);   // V (bf16)
  attn_kernel<<<agrid, gblk, ashm, stream>>>(qb, kb, vb, mask, ctxb, nullptr);
  gemm(ctxb, w_sao, sa_o_b, tmp0, nullptr, MTOK, E_DIM, E_DIM, 0);     // O proj (f32)
  add_ln_kernel<<<MTOK, 256, 0, stream>>>(tmp0, state, ln1_g, ln1_b, x32, xb);

  // ---------------- cross-attention block ----------------
  gemm(xb,     w_caq, ca_q_b, nullptr, qb, MTOK, E_DIM, E_DIM, 0);     // Q from x
  gemm(enc_bf, w_cak, ca_k_b, nullptr, kb, MTOK, E_DIM, E_DIM, 0);     // K from encoder
  gemm(enc_bf, w_cav, ca_v_b, nullptr, vb, MTOK, E_DIM, E_DIM, 0);     // V from encoder
  attn_kernel<<<agrid, gblk, ashm, stream>>>(qb, kb, vb, nullptr, ctxb, out_attn);
  gemm(ctxb, w_cao, ca_o_b, tmp0, nullptr, MTOK, E_DIM, E_DIM, 0);     // O proj (f32)
  add_ln_kernel<<<MTOK, 256, 0, stream>>>(tmp0, x32, ln2_g, ln2_b, y32, yb);

  // ---------------- FFN block ----------------
  gemm(yb,  w_f1, fc1_b, nullptr, f1b, MTOK, FF_DIM, E_DIM, 1);        // relu(fc1), bf16
  gemm(f1b, w_f2, fc2_b, tmp0, nullptr, MTOK, E_DIM, FF_DIM, 0);       // fc2 (f32)
  add_ln_kernel<<<MTOK, 256, 0, stream>>>(tmp0, y32, ln3_g, ln3_b, out_z, nullptr);
}
